// ModalityImputationNet_47184510714458
// MI455X (gfx1250) — compile-verified, matrix operand order checked by reading, not run on hardware
//
#include <hip/hip_runtime.h>

typedef __attribute__((ext_vector_type(16))) __bf16        v16bf;
typedef __attribute__((ext_vector_type(8)))  __bf16        v8bf;
typedef __attribute__((ext_vector_type(2)))  __bf16        v2bf;
typedef __attribute__((ext_vector_type(8)))  float         v8f;
typedef __attribute__((ext_vector_type(4)))  float         f32x4;
typedef __attribute__((ext_vector_type(4)))  unsigned int  u32x4;

#define LDS_STRIDE 40   // bf16 elems per row = 80B: 16B aligned, conflict-free for b128

// f32 pair -> packed bf16 dword (prefer the HW packed convert).
#if __has_builtin(__builtin_amdgcn_cvt_pk_bf16_f32)
__device__ __forceinline__ unsigned int pk_bf16(float a, float b) {
  v2bf v = __builtin_amdgcn_cvt_pk_bf16_f32(a, b);
  return __builtin_bit_cast(unsigned int, v);
}
#else
__device__ __forceinline__ unsigned int pk_bf16(float a, float b) {
  v2bf v;
  v[0] = (__bf16)a;
  v[1] = (__bf16)b;
  return __builtin_bit_cast(unsigned int, v);
}
#endif

// Load a 16x32 (MxK) bf16 WMMA fragment from a row-major LDS tile.
// CDNA5 16-bit A/B layout: lane%16 = row, lanes 0-15 hold K 0..7 & 16..23,
// lanes 16-31 hold K 8..15 & 24..31  -> two 16B ds loads per lane.
__device__ __forceinline__ v16bf load_frag(const __bf16* base, int lane) {
  const int r  = lane & 15;
  const int kb = (lane >> 4) << 3;           // 0 or 8
  const v8bf lo = *(const v8bf*)(base + r * LDS_STRIDE + kb);
  const v8bf hi = *(const v8bf*)(base + r * LDS_STRIDE + kb + 16);
  v16bf out;
#pragma unroll
  for (int i = 0; i < 8; ++i) { out[i] = lo[i]; out[i + 8] = hi[i]; }
  return out;
}

__device__ __forceinline__ v8f wmma_bf16(v16bf a, v16bf b, v8f c) {
  return __builtin_amdgcn_wmma_f32_16x16x32_bf16(false, a, false, b,
                                                 (short)0, c, false, false);
}

template <int NV>
__device__ __forceinline__ void g2r(const float* src, f32x4 r[NV]) {
  const f32x4* p = (const f32x4*)src;
#pragma unroll
  for (int i = 0; i < NV; ++i) r[i] = p[i];
}

// Convert 4*NV f32 -> bf16 and store NV/2 x 16B to LDS.
template <int NV>
__device__ __forceinline__ void r2lds(__bf16* dst, const f32x4 r[NV]) {
#pragma unroll
  for (int i = 0; i < NV / 2; ++i) {
    u32x4 w;
    w[0] = pk_bf16(r[2 * i][0],     r[2 * i][1]);
    w[1] = pk_bf16(r[2 * i][2],     r[2 * i][3]);
    w[2] = pk_bf16(r[2 * i + 1][0], r[2 * i + 1][1]);
    w[3] = pk_bf16(r[2 * i + 1][2], r[2 * i + 1][3]);
    ((u32x4*)dst)[i] = w;
  }
}

// One fused kernel: out[:, z, :] = select(mask, X_{1-z} @ W_z^T + b_z, X_z)
// Block tile 256(M) x 128(N), 8 waves as 4Mx2N, wave tile 64x64:
// per K=32 step each wave does 8 fragment loads (16 ds_load_b128) + 16 WMMAs.
__global__ __launch_bounds__(256)
void modality_impute_wmma(const float* __restrict__ feats,
                          const int*   __restrict__ mask,
                          const float* __restrict__ Wb2c,
                          const float* __restrict__ bb2c,
                          const float* __restrict__ Wc2b,
                          const float* __restrict__ bc2b,
                          float* __restrict__ out,
                          int D, int ld /* = 2*D */) {
  const int slot = blockIdx.z;                  // output modality slot
  const float* W    = (slot == 0) ? Wc2b : Wb2c;
  const float* bias = (slot == 0) ? bc2b : bb2c;
  const float* Ain  = feats + (size_t)(1 - slot) * D;  // GEMM input (other modality)
  const float* pass = feats + (size_t)slot * D;        // passthrough
  float*       outp = out   + (size_t)slot * D;

  __shared__ __bf16 sA[2][256 * LDS_STRIDE];
  __shared__ __bf16 sB[2][128 * LDS_STRIDE];

  const int tid  = threadIdx.x;
  const int lane = tid & 31;
  const int wid  = tid >> 5;
  const int waveM = wid & 3;       // 0..3 : 64-row strip
  const int waveN = wid >> 2;      // 0..1 : 64-col strip

  // ---- cooperative tile loaders ----
  // A: one full 32-float row per thread (256 rows).
  const float* aSrc = Ain + (size_t)(blockIdx.y * 256 + tid) * ld;
  const int aOff = tid * LDS_STRIDE;
  // B: half row (16 floats) per thread (128 rows).
  const int lr = tid >> 1;
  const int lh = tid & 1;
  const float* bSrc = W + (size_t)(blockIdx.x * 128 + lr) * D + lh * 16;
  const int bOff = lr * LDS_STRIDE + lh * 16;

  v8f acc[4][4];
  const v8f vzero = {0.f, 0.f, 0.f, 0.f, 0.f, 0.f, 0.f, 0.f};
#pragma unroll
  for (int m = 0; m < 4; ++m)
#pragma unroll
    for (int n = 0; n < 4; ++n) acc[m][n] = vzero;

  const int kIter = D / 32;

  // ---- prologue: stage 0 ----
  f32x4 ra[8], rb[4];
  g2r<8>(aSrc, ra);
  g2r<4>(bSrc, rb);
  r2lds<8>(&sA[0][aOff], ra);
  r2lds<4>(&sB[0][bOff], rb);
  __syncthreads();

  int buf = 0;
#pragma unroll 1
  for (int kk = 0; kk < kIter; ++kk) {
    const bool hasNext = (kk + 1 < kIter);
    if (hasNext) {
      g2r<8>(aSrc + (kk + 1) * 32, ra);
      g2r<4>(bSrc + (kk + 1) * 32, rb);
    }

    // ---- compute from LDS[buf]: 8 fragment loads, 16 WMMAs ----
    const __bf16* sAb = &sA[buf][0];
    const __bf16* sBb = &sB[buf][0];
    v16bf nbf[4];
#pragma unroll
    for (int nn = 0; nn < 4; ++nn)
      nbf[nn] = load_frag(sBb + (waveN * 64 + nn * 16) * LDS_STRIDE, lane);
#pragma unroll
    for (int m = 0; m < 4; ++m) {
      v16bf af = load_frag(sAb + (waveM * 64 + m * 16) * LDS_STRIDE, lane);
#pragma unroll
      for (int nn = 0; nn < 4; ++nn)
        acc[m][nn] = wmma_bf16(af, nbf[nn], acc[m][nn]);
    }

    if (hasNext) {
      r2lds<8>(&sA[buf ^ 1][aOff], ra);
      r2lds<4>(&sB[buf ^ 1][bOff], rb);
      __syncthreads();
      buf ^= 1;
    }
  }

  // ---- epilogue: bias + mask select + store ----
  // C/D layout: lane%16 = N, lanes 0-15 rows v, lanes 16-31 rows v+8.
  const int mBase = blockIdx.y * 256 + waveM * 64;
  const int nBase = blockIdx.x * 128 + waveN * 64;
  const int rsub  = (lane >> 4) << 3;   // 0 or 8
  const int ncol  = lane & 15;

#pragma unroll
  for (int m = 0; m < 4; ++m) {
    const int row0 = mBase + m * 16 + rsub;
    bool need[8];
#pragma unroll
    for (int v = 0; v < 8; ++v) {
      const int r  = row0 + v;
      const int m0 = mask[2 * r + 0];
      const int m1 = mask[2 * r + 1];
      need[v] = (slot == 0) ? (m0 == 0 && m1 == 1) : (m0 == 1 && m1 == 0);
    }
#pragma unroll
    for (int nn = 0; nn < 4; ++nn) {
      const int col = nBase + nn * 16 + ncol;
      const float bv = bias[col];
#pragma unroll
      for (int v = 0; v < 8; ++v) {
        const size_t idx = (size_t)(row0 + v) * ld + col;
        const float val = acc[m][nn][v] + bv;
        outp[idx] = need[v] ? val : pass[idx];
      }
    }
  }
}

extern "C" void kernel_launch(void* const* d_in, const int* in_sizes, int n_in,
                              void* d_out, int out_size, void* d_ws, size_t ws_size,
                              hipStream_t stream) {
  const float* feats = (const float*)d_in[0];
  const int*   mask  = (const int*)  d_in[1];
  const float* Wb2c  = (const float*)d_in[2];
  const float* bb2c  = (const float*)d_in[3];
  const float* Wc2b  = (const float*)d_in[4];
  const float* bc2b  = (const float*)d_in[5];
  float* out = (float*)d_out;

  const int D = in_sizes[3];               // b_b2c length = 1024
  const int B = in_sizes[0] / (2 * D);     // 16384

  dim3 grid(D / 128, B / 256, 2);
  modality_impute_wmma<<<grid, 256, 0, stream>>>(feats, mask, Wb2c, bb2c,
                                                 Wc2b, bc2b, out, D, 2 * D);
}
